// GCN_49460843381579
// MI455X (gfx1250) — matrix operand order repaired and checked by reading, not compiled, not run on hardware
//
#include <hip/hip_runtime.h>
#include <math.h>

#define NN 100000
#define EE 1600000
#define GG 512
#define HIDF 64
#define F3 768
#define CLSN 10

typedef __attribute__((ext_vector_type(16))) _Float16 v16h;
typedef __attribute__((ext_vector_type(8)))  float    v8f;

union H16 { v16h v; uint4 u[2]; };

static inline int cdiv(int a, int b) { return (a + b - 1) / b; }

// ---------------- utility kernels ----------------

__global__ void fill_f32(float* __restrict__ p, float v, int n) {
  int i = blockIdx.x * blockDim.x + threadIdx.x;
  if (i < n) p[i] = v;
}

__global__ void cvt_f32_f16(const float* __restrict__ src, _Float16* __restrict__ dst, int n) {
  int i = blockIdx.x * blockDim.x + threadIdx.x;
  if (i < n) dst[i] = (_Float16)src[i];
}

// Pack W[K=64][ncols] (f32, row-major) into WMMA B-operand layout (f16).
// Layout: [tile t][kseg s(2)][lane(32)][16 halves]; lane: N = 16t + lane%16,
// K = s*32 + (lane/16)*16 + j, j = 0..15 contiguous.
__global__ void pack_b(const float* __restrict__ W, _Float16* __restrict__ out, int ncols, int ntiles) {
  int tid = blockIdx.x * blockDim.x + threadIdx.x;
  if (tid >= ntiles * 64) return;
  int t = tid >> 6;
  int rem = tid & 63;
  int s = rem >> 5;
  int lane = rem & 31;
  int n = t * 16 + (lane & 15);
  int kbase = s * 32 + (lane >> 4) * 16;
  _Float16* o = out + ((size_t)(t * 2 + s) * 32 + lane) * 16;
#pragma unroll
  for (int j = 0; j < 16; ++j) o[j] = (_Float16)W[(size_t)(kbase + j) * ncols + n];
}

// ---------------- WMMA GEMM: [rows x 64] (f16) @ Bpacked -> [rows x ncols] f32 ----------------
// One wave per (16-row strip, 16-col tile). K = 64 -> two v_wmma_f32_16x16x32_f16.
__global__ __launch_bounds__(256) void gemm16(const _Float16* __restrict__ A,
                                              const _Float16* __restrict__ Bp,
                                              float* __restrict__ D,
                                              int nstrips, int ntiles, int ncols) {
  int w    = (blockIdx.x * blockDim.x + threadIdx.x) >> 5;
  int lane = threadIdx.x & 31;
  if (w >= nstrips * ntiles) return;   // wave-uniform: EXEC stays all-ones for active waves
  int strip = w / ntiles;
  int t     = w % ntiles;

  // A operand (contiguous K order, matching packed B): lane<16 -> K base 0, lane>=16 -> base 16;
  // halves 0..15 = K[klo..klo+15]  (contiguous 16B runs)
  int row = strip * 16 + (lane & 15);
  int klo = (lane >> 4) * 16;
  const _Float16* ar = A + (size_t)row * 64;
  H16 a0, a1, b0, b1;
  a0.u[0] = *(const uint4*)(ar + klo);
  a0.u[1] = *(const uint4*)(ar + klo + 8);
  a1.u[0] = *(const uint4*)(ar + 32 + klo);
  a1.u[1] = *(const uint4*)(ar + 32 + klo + 8);

  const _Float16* bb = Bp + ((size_t)(t * 2) * 32 + lane) * 16;
  b0.u[0] = *(const uint4*)(bb);
  b0.u[1] = *(const uint4*)(bb + 8);
  b1.u[0] = *(const uint4*)(bb + 32 * 16);
  b1.u[1] = *(const uint4*)(bb + 32 * 16 + 8);

  v8f acc = {};
  acc = __builtin_amdgcn_wmma_f32_16x16x32_f16(false, a0.v, false, b0.v, (short)0, acc, false, false);
  acc = __builtin_amdgcn_wmma_f32_16x16x32_f16(false, a1.v, false, b1.v, (short)0, acc, false, false);

  // D layout: VGPR v -> M = v + (lane/16)*8 ; N = lane%16
  int m0 = strip * 16 + (lane >> 4) * 8;
  int n  = t * 16 + (lane & 15);
#pragma unroll
  for (int v = 0; v < 8; ++v) D[(size_t)(m0 + v) * ncols + n] = acc[v];
}

// ---------------- graph kernels ----------------

__global__ void deg_kernel(float* __restrict__ deg, const int* __restrict__ dst, int E) {
  int i = blockIdx.x * blockDim.x + threadIdx.x;
  if (i < E) atomicAdd(&deg[dst[i]], 1.0f);
}

__global__ void dinv_kernel(const float* __restrict__ deg, float* __restrict__ d1,
                            float* __restrict__ d2, int n) {
  int i = blockIdx.x * blockDim.x + threadIdx.x;
  if (i >= n) return;
  float d = deg[i];
  d1[i] = rsqrtf(d + 2.0f);
  d2[i] = rsqrtf(d + 1.0f);
}

// agg[i][f] = fill*dinv[i]^2*hW[i][f] + b[f]   (self-loop + bias, also zero-inits agg)
__global__ void selfinit(float* __restrict__ agg, const float* __restrict__ hW,
                         const float* __restrict__ dinv, const float* __restrict__ b,
                         float fill, int total) {
  int i = blockIdx.x * blockDim.x + threadIdx.x;
  if (i >= total) return;
  int node = i >> 6;
  int f = i & 63;
  float di = dinv[node];
  agg[i] = fill * di * di * hW[i] + b[f];
}

// 8 threads per edge; each handles 8 contiguous features: agg[dst] += hW[src]*norm
__global__ void scatter_edges(float* __restrict__ agg, const float* __restrict__ hW,
                              const float* __restrict__ dinv, const int* __restrict__ src,
                              const int* __restrict__ dst, int E) {
  int tid = blockIdx.x * blockDim.x + threadIdx.x;
  if (tid >= E * 8) return;
  int e = tid >> 3;
  int g = (tid & 7) * 8;
  int s = src[e], d = dst[e];
  float nrm = dinv[s] * dinv[d];
  const float* hs = hW + (size_t)s * 64 + g;
  float4 v0 = *(const float4*)(hs);
  float4 v1 = *(const float4*)(hs + 4);
  float* o = agg + (size_t)d * 64 + g;
  atomicAdd(o + 0, v0.x * nrm); atomicAdd(o + 1, v0.y * nrm);
  atomicAdd(o + 2, v0.z * nrm); atomicAdd(o + 3, v0.w * nrm);
  atomicAdd(o + 4, v1.x * nrm); atomicAdd(o + 5, v1.y * nrm);
  atomicAdd(o + 6, v1.z * nrm); atomicAdd(o + 7, v1.w * nrm);
}

// LayerNorm(64) + LeakyReLU(0.2), one wave32 per row, in place
__global__ void ln_leaky64(float* __restrict__ h, const float* __restrict__ g,
                           const float* __restrict__ b, int nrows) {
  int wid  = (blockIdx.x * blockDim.x + threadIdx.x) >> 5;
  int lane = threadIdx.x & 31;
  if (wid >= nrows) return;
  float* row = h + (size_t)wid * 64;
  float x0 = row[lane], x1 = row[lane + 32];
  float s = x0 + x1, sq = x0 * x0 + x1 * x1;
#pragma unroll
  for (int m = 16; m; m >>= 1) { s += __shfl_xor(s, m, 32); sq += __shfl_xor(sq, m, 32); }
  float mu  = s * (1.0f / 64.0f);
  float var = sq * (1.0f / 64.0f) - mu * mu;
  float inv = rsqrtf(var + 1e-5f);
  float y0 = (x0 - mu) * inv * g[lane] + b[lane];
  float y1 = (x1 - mu) * inv * g[lane + 32] + b[lane + 32];
  row[lane]      = y0 > 0.0f ? y0 : 0.2f * y0;
  row[lane + 32] = y1 > 0.0f ? y1 : 0.2f * y1;
}

__device__ inline void atomicMaxF(float* a, float v) {
  if (v >= 0.0f) atomicMax((int*)a, __float_as_int(v));
  else           atomicMin((unsigned int*)a, __float_as_uint(v));
}

// LeakyReLU then segment_max into pooled[batch[node]][f]
__global__ void leaky_pool(const float* __restrict__ agg, const int* __restrict__ batch,
                           float* __restrict__ pooled, int total) {
  int i = blockIdx.x * blockDim.x + threadIdx.x;
  if (i >= total) return;
  int node = i >> 6;
  int f = i & 63;
  float v = agg[i];
  v = v > 0.0f ? v : 0.2f * v;
  atomicMaxF(&pooled[(size_t)batch[node] * 64 + f], v);
}

// z[row] = LeakyReLU(LN(z[row] + bias)) over 768 features; one block per row
__global__ void ln_leaky768(float* __restrict__ z, const float* __restrict__ bias,
                            const float* __restrict__ g, const float* __restrict__ b) {
  int row = blockIdx.x, tid = threadIdx.x;
  float* zr = z + (size_t)row * 768;
  float v[3], s = 0.0f, sq = 0.0f;
#pragma unroll
  for (int i = 0; i < 3; ++i) {
    float x = zr[tid + i * 256] + bias[tid + i * 256];
    v[i] = x; s += x; sq += x * x;
  }
  __shared__ float rs[256], rq[256];
  rs[tid] = s; rq[tid] = sq; __syncthreads();
  for (int m = 128; m; m >>= 1) {
    if (tid < m) { rs[tid] += rs[tid + m]; rq[tid] += rq[tid + m]; }
    __syncthreads();
  }
  float mu  = rs[0] * (1.0f / 768.0f);
  float var = rq[0] * (1.0f / 768.0f) - mu * mu;
  float inv = rsqrtf(var + 1e-5f);
#pragma unroll
  for (int i = 0; i < 3; ++i) {
    int f = tid + i * 256;
    float y = (v[i] - mu) * inv * g[f] + b[f];
    zr[f] = y > 0.0f ? y : 0.2f * y;
  }
}

// logits = z @ W4 + b4 ; softmax ; one block per graph
__global__ void head_kernel(const float* __restrict__ z, const float* __restrict__ W4,
                            const float* __restrict__ b4, float* __restrict__ out) {
  int gi = blockIdx.x, tid = threadIdx.x;
  const float* zr = z + (size_t)gi * 768;
  float acc[CLSN];
#pragma unroll
  for (int c = 0; c < CLSN; ++c) acc[c] = 0.0f;
  for (int k = tid; k < 768; k += 256) {
    float x = zr[k];
    const float* wr = W4 + (size_t)k * CLSN;
#pragma unroll
    for (int c = 0; c < CLSN; ++c) acc[c] += x * wr[c];
  }
  __shared__ float red[256];
  __shared__ float logit[CLSN];
  for (int c = 0; c < CLSN; ++c) {
    red[tid] = acc[c]; __syncthreads();
    for (int m = 128; m; m >>= 1) {
      if (tid < m) red[tid] += red[tid + m];
      __syncthreads();
    }
    if (tid == 0) logit[c] = red[0] + b4[c];
    __syncthreads();
  }
  if (tid == 0) {
    float mx = logit[0];
#pragma unroll
    for (int c = 1; c < CLSN; ++c) mx = fmaxf(mx, logit[c]);
    float e[CLSN], ssum = 0.0f;
#pragma unroll
    for (int c = 0; c < CLSN; ++c) { e[c] = expf(logit[c] - mx); ssum += e[c]; }
    float is = 1.0f / ssum;
#pragma unroll
    for (int c = 0; c < CLSN; ++c) out[(size_t)gi * CLSN + c] = e[c] * is;
  }
}

// ---------------- launcher ----------------

extern "C" void kernel_launch(void* const* d_in, const int* in_sizes, int n_in,
                              void* d_out, int out_size, void* d_ws, size_t ws_size,
                              hipStream_t stream) {
  const float* x     = (const float*)d_in[0];
  const int*   ei    = (const int*)d_in[1];
  const int*   batch = (const int*)d_in[2];
  const float* W1 = (const float*)d_in[3];
  const float* b1 = (const float*)d_in[4];
  const float* lng = (const float*)d_in[5];
  const float* lnb = (const float*)d_in[6];
  const float* W2 = (const float*)d_in[7];
  const float* b2 = (const float*)d_in[8];
  const float* W3 = (const float*)d_in[9];
  const float* b3 = (const float*)d_in[10];
  const float* ln2g = (const float*)d_in[11];
  const float* ln2b = (const float*)d_in[12];
  const float* W4 = (const float*)d_in[13];
  const float* b4 = (const float*)d_in[14];
  const int* src = ei;
  const int* dst = ei + EE;

  // workspace carve (all chunks keep 16B alignment)
  float* deg    = (float*)d_ws;
  float* dinv1  = deg + NN;
  float* dinv2  = dinv1 + NN;
  float* buf0   = dinv2 + NN;              // hW  [N,64]
  float* buf1   = buf0 + (size_t)NN * 64;  // agg/h [N,64]
  float* pooled = buf1 + (size_t)NN * 64;  // [G,64]
  float* z      = pooled + (size_t)GG * 64;  // [G,768]
  _Float16* hf  = (_Float16*)(z + (size_t)GG * 768);  // f16 activations [N,64]
  _Float16* w1p = hf + (size_t)NN * 64;    // 4096 halves
  _Float16* w2p = w1p + 4096;
  _Float16* w3p = w2p + 4096;              // 49152 halves

  const int B = 256;

  // degrees + normalization
  fill_f32<<<cdiv(NN, B), B, 0, stream>>>(deg, 0.0f, NN);
  deg_kernel<<<cdiv(EE, B), B, 0, stream>>>(deg, dst, EE);
  dinv_kernel<<<cdiv(NN, B), B, 0, stream>>>(deg, dinv1, dinv2, NN);

  // pack weights into WMMA B layout
  pack_b<<<1, B, 0, stream>>>(W1, w1p, 64, 4);
  pack_b<<<1, B, 0, stream>>>(W2, w2p, 64, 4);
  pack_b<<<12, B, 0, stream>>>(W3, w3p, 768, 48);

  // ---- layer 1 ----
  cvt_f32_f16<<<cdiv(NN * 64, B), B, 0, stream>>>(x, hf, NN * 64);
  gemm16<<<cdiv((NN / 16) * 4 * 32, B), B, 0, stream>>>(hf, w1p, buf0, NN / 16, 4, 64);
  selfinit<<<cdiv(NN * 64, B), B, 0, stream>>>(buf1, buf0, dinv1, b1, 2.0f, NN * 64);
  scatter_edges<<<cdiv(EE * 8, B), B, 0, stream>>>(buf1, buf0, dinv1, src, dst, EE);
  ln_leaky64<<<cdiv(NN * 32, B), B, 0, stream>>>(buf1, lng, lnb, NN);

  // ---- layer 2 ----
  cvt_f32_f16<<<cdiv(NN * 64, B), B, 0, stream>>>(buf1, hf, NN * 64);
  gemm16<<<cdiv((NN / 16) * 4 * 32, B), B, 0, stream>>>(hf, w2p, buf0, NN / 16, 4, 64);
  selfinit<<<cdiv(NN * 64, B), B, 0, stream>>>(buf1, buf0, dinv2, b2, 1.0f, NN * 64);
  scatter_edges<<<cdiv(EE * 8, B), B, 0, stream>>>(buf1, buf0, dinv2, src, dst, EE);

  // ---- pooling + head ----
  fill_f32<<<cdiv(GG * 64, B), B, 0, stream>>>(pooled, -__builtin_huge_valf(), GG * 64);
  leaky_pool<<<cdiv(NN * 64, B), B, 0, stream>>>(buf1, batch, pooled, NN * 64);
  cvt_f32_f16<<<cdiv(GG * 64, B), B, 0, stream>>>(pooled, hf, GG * 64);
  gemm16<<<cdiv((GG / 16) * 48 * 32, B), B, 0, stream>>>(hf, w3p, z, GG / 16, 48, 768);
  ln_leaky768<<<GG, B, 0, stream>>>(z, b3, ln2g, ln2b);
  head_kernel<<<GG, B, 0, stream>>>(z, W4, b4, (float*)d_out);
}
